// CTrans_42133629173960
// MI455X (gfx1250) — matrix operand order repaired and checked by reading, not compiled, not run on hardware
//
#include <hip/hip_runtime.h>
#include <hip/hip_bf16.h>

// ---------------------------------------------------------------------------
// CDNA5 (gfx1250) Convpass transformer block.
// All large GEMMs + attention matmuls via v_wmma_f32_16x16x32_bf16 (wave32).
// Weights pre-transposed so ALL WMMA fragments load as contiguous ds_load_b128.
// GEMM tiles staged with gfx1250 async global->LDS loads (ASYNCcnt), double
// buffered so WMMA on tile k overlaps the async fill of tile k+1.
// ---------------------------------------------------------------------------

typedef __attribute__((ext_vector_type(16))) __bf16 v16bf;
typedef __attribute__((ext_vector_type(8)))  float  v8f;
typedef int v4i __attribute__((vector_size(16)));   // matches async-LDS builtin

#define KB   32          // batch
#define KN   512         // tokens
#define KC   512         // channels
#define KH   8           // heads
#define KDH  64          // head dim
#define KADIM 8
#define KMLP 4096
#define KM   (KB * KN)   // 16384 GEMM rows

// ---- gfx1250 async global->LDS (guarded: falls back to plain copies) ------
#if defined(__has_builtin)
#if __has_builtin(__builtin_amdgcn_global_load_async_to_lds_b128) && \
    __has_builtin(__builtin_amdgcn_s_wait_asynccnt)
#define HAVE_ASYNC 1
#endif
#endif
#ifndef HAVE_ASYNC
#define HAVE_ASYNC 0
#endif

__device__ __forceinline__ void async_cp16(void* l, const void* g) {
#if HAVE_ASYNC
    __builtin_amdgcn_global_load_async_to_lds_b128(
        (__attribute__((address_space(1))) v4i*)g,
        (__attribute__((address_space(3))) v4i*)l,
        /*offset=*/0, /*cpol=*/0);
#else
    *(uint4*)l = *(const uint4*)g;
#endif
}
#if HAVE_ASYNC
#define WAIT_ASYNC(n) __builtin_amdgcn_s_wait_asynccnt(n)
#else
#define WAIT_ASYNC(n) ((void)0)
#endif

__device__ __forceinline__ float qgelu_f(float x) {
    return x / (1.0f + __expf(-1.702f * x));
}
__device__ __forceinline__ float gelu_erf_f(float x) {
    return 0.5f * x * (1.0f + erff(x * 0.70710678118654752f));
}

// --------------------------- WMMA fragment loaders -------------------------
// A tile [M x K] row-major (ld multiple of 8 -> 16B-aligned rows).
// 16-bit A 16x32 layout (ISA 7.12.2): lane = M; element pairs map K with 8*half
// offset -> per lane two contiguous 16B runs (2x ds_load_b128).
__device__ __forceinline__ v16bf frag_a16(const __bf16* base, int ld, int lane) {
    const int m = lane & 15, half = lane >> 4;
    v16bf f;
#pragma unroll
    for (int g = 0; g < 8; ++g) {
        const int kb = (g < 4) ? (2 * g + 8 * half) : (16 + 2 * (g - 4) + 8 * half);
        f[2 * g]     = base[m * ld + kb];
        f[2 * g + 1] = base[m * ld + kb + 1];
    }
    return f;
}
// B fragment from B^T stored row-major [N x K]: lane = n, element=k with 8*half
// offset (same K order as A) -> two contiguous 16B runs per lane (2x ds_load_b128).
__device__ __forceinline__ v16bf frag_bT16(const __bf16* base, int ld, int lane) {
    const int n = lane & 15, half = lane >> 4;
    v16bf f;
#pragma unroll
    for (int e = 0; e < 16; ++e)
        f[e] = base[n * ld + ((e < 8) ? (8 * half + e) : (16 + 8 * half + (e - 8)))];
    return f;
}

__device__ __forceinline__ v8f wmma_bf16(v16bf a, v16bf b, v8f c) {
    return __builtin_amdgcn_wmma_f32_16x16x32_bf16(
        /*neg_a=*/false, a, /*neg_b=*/false, b,
        /*c_mod=*/(short)0, c, /*reuse_a=*/false, /*reuse_b=*/false);
}

// ------------------------------- GEMM kernel -------------------------------
// C[M,N] = act(A[M,K](bf16) * W[K,N] + bias), W given TRANSPOSED: WT[N,K] bf16.
// Block tile 128x128 (8 waves, wave tile 64x32 = 4x2 WMMA accums), K-step 32.
// 2-deep async pipeline: 4 async b128 instructions per tile per wave.
// ACT: 0=none 1=gelu(erf) 2=qgelu.  ACCUM: outF += v.  WB16: store bf16 copy.
#define GT_M 128
#define GT_N 128
#define GT_K 32
#define LDT  40          // padded LDS row stride (bf16 elems): 80B, conflict-free

template <int ACT, bool ACCUM, bool WF32, bool WB16>
__global__ __launch_bounds__(256) void gemm_wmma_kernel(
    const __bf16* __restrict__ A, const __bf16* __restrict__ WT,
    const float* __restrict__ bias, float* __restrict__ outF,
    __bf16* __restrict__ outB, int M, int N, int K)
{
    __shared__ alignas(16) __bf16 As [2][GT_M][LDT];   // 2 x 10 KB
    __shared__ alignas(16) __bf16 BTs[2][GT_N][LDT];   // 2 x 10 KB

    const int tid  = threadIdx.x;
    const int lane = tid & 31;
    const int wave = tid >> 5;
    const int wm = wave >> 2, wn = wave & 3;   // 2x4 wave grid -> 64x32 per wave
    const int m0 = blockIdx.y * GT_M;
    const int n0 = blockIdx.x * GT_N;
    const int mlane = lane & 15, half = lane >> 4;

    v8f acc[4][2] = {};

    // stage one K-tile (128 rows x 32 elems = 512 x 16B chunks; 2 per thread
    // for A and for B^T -> 4 async instructions per wave per tile)
    auto stage = [&](int k0, int buf) {
#pragma unroll
        for (int i = 0; i < 2; ++i) {
            const int ci = tid + i * 256;
            const int r = ci >> 2, c = (ci & 3) * 8;
            async_cp16(&As[buf][r][c],  &A [(size_t)(m0 + r) * K + (k0 + c)]);
            async_cp16(&BTs[buf][r][c], &WT[(size_t)(n0 + r) * K + (k0 + c)]);
        }
    };

    const int nk = K / GT_K;
    stage(0, 0);
    stage(GT_K, 1);            // nk >= 16 for every GEMM in this model

    for (int kt = 0; kt < nk; ++kt) {
        // tile kt done when <= 4 async ops (the next tile) remain outstanding
        if (kt + 1 < nk) WAIT_ASYNC(4); else WAIT_ASYNC(0);
        __syncthreads();

        const int buf = kt & 1;
        v16bf af[4], bfm[2];
#pragma unroll
        for (int i = 0; i < 4; ++i) af[i]  = frag_a16 (&As [buf][wm * 64 + i * 16][0], LDT, lane);
#pragma unroll
        for (int j = 0; j < 2; ++j) bfm[j] = frag_bT16(&BTs[buf][wn * 32 + j * 16][0], LDT, lane);
#pragma unroll
        for (int i = 0; i < 4; ++i)
#pragma unroll
            for (int j = 0; j < 2; ++j)
                acc[i][j] = wmma_bf16(af[i], bfm[j], acc[i][j]);
        __syncthreads();       // all waves done reading buf before refilling it

        if (kt + 2 < nk) stage((kt + 2) * GT_K, buf);
    }

    // epilogue: C/D layout -> VGPR r: M = r + 8*half, N = mlane
#pragma unroll
    for (int i = 0; i < 4; ++i) {
#pragma unroll
        for (int j = 0; j < 2; ++j) {
            const int rowb = m0 + wm * 64 + i * 16 + 8 * half;
            const int col  = n0 + wn * 32 + j * 16 + mlane;
#pragma unroll
            for (int r = 0; r < 8; ++r) {
                float v = acc[i][j][r];
                if (bias) v += bias[col];
                if (ACT == 1) v = gelu_erf_f(v);
                else if (ACT == 2) v = qgelu_f(v);
                const size_t idx = (size_t)(rowb + r) * N + col;
                if (WF32) { if (ACCUM) outF[idx] += v; else outF[idx] = v; }
                if (WB16) outB[idx] = (__bf16)v;
            }
        }
    }
}

// --------------------------- Flash attention kernel ------------------------
// grid = (B*H, N/64); block = 256 (8 waves). Online-softmax flash attention.
// qkv (bf16) layout [B*N, 3*C]; q col = h*64, k col = 512+h*64, v col = 1024+h*64.
#define LDA  72   // padded stride for 64-wide bf16 tiles (144B rows, 16B aligned)

__global__ __launch_bounds__(256) void flash_attn_kernel(
    const __bf16* __restrict__ qkv, __bf16* __restrict__ o)
{
    const int bh = blockIdx.x;
    const int b = bh >> 3, h = bh & 7;
    const int qb = blockIdx.y;
    const int tid = threadIdx.x, lane = tid & 31, wave = tid >> 5;
    const int mlane = lane & 15, half = lane >> 4;

    __shared__ alignas(16) __bf16 Qs [64][LDA];   // 9 KB (unscaled)
    __shared__ alignas(16) __bf16 Ks [64][LDA];   // 9 KB
    __shared__ alignas(16) __bf16 VTs[64][LDA];   // 9 KB (transposed: [d][key])
    __shared__ alignas(16) __bf16 Ps [64][LDA];   // 9 KB
    __shared__ float  Ss[64][68];                 // 17 KB, padded vs bank conflicts
    __shared__ float  mrow[64], lrow[64], srow[64];

    const size_t rowbase = (size_t)b * KN;
    const int qoff = h * 64, koff = KC + h * 64, voff = 2 * KC + h * 64;
    const float scale = 0.125f;  // 64^-0.5 applied to raw scores

    // stage Q via async: 64x64 = 512 x 16B chunks, 2 per thread
#pragma unroll
    for (int i = 0; i < 2; ++i) {
        const int ci = tid + i * 256;
        const int r = ci >> 3, c = (ci & 7) * 8;
        async_cp16(&Qs[r][c],
                   &qkv[(rowbase + qb * 64 + r) * (3 * KC) + qoff + c]);
    }
    if (tid < 64) { mrow[tid] = -3.0e38f; lrow[tid] = 0.0f; }

    const int t0 = wave * 2;   // each wave owns 2 of the 16 (mi,ni) 16x16 tiles
    v8f accO[2] = {};

    for (int j = 0; j < 8; ++j) {
        __syncthreads();
        // K tile: async straight copy (used as B^T); V tile: transposed scatter
#pragma unroll
        for (int i = 0; i < 2; ++i) {
            const int ci = tid + i * 256;
            const int r = ci >> 3, c = (ci & 7) * 8;
            async_cp16(&Ks[r][c],
                       &qkv[(rowbase + j * 64 + r) * (3 * KC) + koff + c]);
        }
        for (int e = tid; e < 4096; e += 256) {
            const int r = e >> 6, c = e & 63;
            VTs[c][r] = qkv[(rowbase + j * 64 + r) * (3 * KC) + voff + c];
        }
        WAIT_ASYNC(0);
        __syncthreads();

        // S tile = Q(64x64) * K^T(64x64): 2 WMMAs per 16x16 tile
#pragma unroll
        for (int t = 0; t < 2; ++t) {
            const int mi = (t0 + t) >> 2, ni = (t0 + t) & 3;
            v8f s = {};
#pragma unroll
            for (int kk = 0; kk < 2; ++kk) {
                v16bf a  = frag_a16 (&Qs[mi * 16][kk * 32], LDA, lane);
                v16bf bt = frag_bT16(&Ks[ni * 16][kk * 32], LDA, lane);
                s = wmma_bf16(a, bt, s);
            }
#pragma unroll
            for (int r = 0; r < 8; ++r)
                Ss[mi * 16 + r + 8 * half][ni * 16 + mlane] = s[r];
        }
        __syncthreads();

        // online softmax per row (threads 0..63); scale folded in here
        if (tid < 64) {
            const float m_old = mrow[tid];
            float mx = m_old;
            for (int c = 0; c < 64; ++c) mx = fmaxf(mx, Ss[tid][c] * scale);
            const float sc = __expf(m_old - mx);
            float sum = 0.0f;
            for (int c = 0; c < 64; ++c) {
                const float ev = __expf(Ss[tid][c] * scale - mx);
                Ps[tid][c] = (__bf16)ev;
                sum += ev;
            }
            lrow[tid] = lrow[tid] * sc + sum;
            mrow[tid] = mx;
            srow[tid] = sc;
        }
        __syncthreads();

        // O = O*sc + P(64x64) * V(64x64); V^T in LDS -> contiguous B frags
#pragma unroll
        for (int t = 0; t < 2; ++t) {
            const int mi = (t0 + t) >> 2, ni = (t0 + t) & 3;
#pragma unroll
            for (int r = 0; r < 8; ++r) accO[t][r] *= srow[mi * 16 + r + 8 * half];
#pragma unroll
            for (int kk = 0; kk < 2; ++kk) {
                v16bf a  = frag_a16 (&Ps[mi * 16][kk * 32], LDA, lane);
                v16bf bb = frag_bT16(&VTs[ni * 16][kk * 32], LDA, lane);
                accO[t] = wmma_bf16(a, bb, accO[t]);
            }
        }
    }

    // epilogue: divide by l, store bf16 for the proj GEMM
#pragma unroll
    for (int t = 0; t < 2; ++t) {
        const int mi = (t0 + t) >> 2, ni = (t0 + t) & 3;
#pragma unroll
        for (int r = 0; r < 8; ++r) {
            const int row = mi * 16 + r + 8 * half;
            const float val = accO[t][r] / lrow[row];
            o[(rowbase + qb * 64 + row) * KC + h * 64 + ni * 16 + mlane] = (__bf16)val;
        }
    }
}

// ------------------------------ small kernels ------------------------------
__global__ __launch_bounds__(256) void add2_kernel(const float* a, const float* b,
                                                   float* o, int n) {
    const int i = blockIdx.x * 256 + threadIdx.x;
    if (i < n) o[i] = a[i] + b[i];
}
__global__ __launch_bounds__(256) void copy_kernel(const float* s, float* d, int n) {
    const int i = blockIdx.x * 256 + threadIdx.x;
    if (i < n) d[i] = s[i];
}
// transpose + convert: WT[n*K + k] = (bf16)W[k*N + n]
__global__ __launch_bounds__(256) void cvt_bf16_t_kernel(const float* __restrict__ s,
                                                         __bf16* __restrict__ d,
                                                         int Kd, int Nd) {
    const int i = blockIdx.x * 256 + threadIdx.x;
    if (i < Kd * Nd) {
        const int k = i / Nd, n = i % Nd;
        d[(size_t)n * Kd + k] = (__bf16)s[i];
    }
}

// LayerNorm: one block per row of 512; writes f32 (for Convpass) + bf16 (for WMMA A)
__global__ __launch_bounds__(256) void ln_kernel(
    const float* __restrict__ x, const float* __restrict__ g,
    const float* __restrict__ bb, float* __restrict__ hf, __bf16* __restrict__ hb)
{
    const int row = blockIdx.x;
    const int tid = threadIdx.x;
    const float* xr = x + (size_t)row * KC;
    __shared__ float red[256];

    float s = 0.0f;
    for (int c = tid; c < KC; c += 256) s += xr[c];
    red[tid] = s; __syncthreads();
    for (int o = 128; o > 0; o >>= 1) { if (tid < o) red[tid] += red[tid + o]; __syncthreads(); }
    const float mean = red[0] * (1.0f / KC);
    __syncthreads();

    float v = 0.0f;
    for (int c = tid; c < KC; c += 256) { const float d = xr[c] - mean; v += d * d; }
    red[tid] = v; __syncthreads();
    for (int o = 128; o > 0; o >>= 1) { if (tid < o) red[tid] += red[tid + o]; __syncthreads(); }
    const float rstd = rsqrtf(red[0] * (1.0f / KC) + 1e-5f);

    for (int c = tid; c < KC; c += 256) {
        const float y = (xr[c] - mean) * rstd * g[c] + bb[c];
        hf[(size_t)row * KC + c] = y;
        hb[(size_t)row * KC + c] = (__bf16)y;
    }
}

// Convpass down: d[row,a] = qgelu(h[row,:] . dw[:,a] + db[a]); one thread per (row,a)
__global__ __launch_bounds__(256) void cp_down_kernel(
    const float* __restrict__ h, const float* __restrict__ dw,
    const float* __restrict__ db, float* __restrict__ d)
{
    const int idx = blockIdx.x * 256 + threadIdx.x;   // KM*8
    if (idx >= KM * KADIM) return;
    const int a = idx & 7, row = idx >> 3;
    float s = db[a];
    for (int c = 0; c < KC; ++c) s += h[(size_t)row * KC + c] * dw[c * KADIM + a];
    d[idx] = qgelu_f(s);
}

// 3x3x3 SAME conv over the 8x8x8 token grid, 8->8 channels (NDHWC storage)
__global__ __launch_bounds__(256) void cp_conv_kernel(
    const float* __restrict__ d, const float* __restrict__ cw,
    const float* __restrict__ cb, float* __restrict__ p)
{
    const int idx = blockIdx.x * 256 + threadIdx.x;   // B*512*8
    if (idx >= KB * KN * KADIM) return;
    const int co = idx & 7;
    const int n  = (idx >> 3) & 511;
    const int b  = idx >> 12;
    const int z = n >> 6, y = (n >> 3) & 7, xw = n & 7;
    float s = cb[co];
    for (int ci = 0; ci < KADIM; ++ci)
        for (int kz = 0; kz < 3; ++kz) {
            const int zz = z + kz - 1; if (zz < 0 || zz > 7) continue;
            for (int ky = 0; ky < 3; ++ky) {
                const int yy = y + ky - 1; if (yy < 0 || yy > 7) continue;
                for (int kx = 0; kx < 3; ++kx) {
                    const int xx = xw + kx - 1; if (xx < 0 || xx > 7) continue;
                    s += d[((size_t)b * KN + zz * 64 + yy * 8 + xx) * KADIM + ci] *
                         cw[((((co * KADIM) + ci) * 3 + kz) * 3 + ky) * 3 + kx];
                }
            }
        }
    p[idx] = s;
}

// Convpass up (fused residual add): out[row,c] += qgelu(p[row,:]) . uw[:,c] + ub[c]
__global__ __launch_bounds__(256) void cp_up_add_kernel(
    const float* __restrict__ p, const float* __restrict__ uw,
    const float* __restrict__ ub, float* __restrict__ out)
{
    const int idx = blockIdx.x * 256 + threadIdx.x;   // KM*KC
    if (idx >= KM * KC) return;
    const int c = idx & 511, row = idx >> 9;
    float s = ub[c];
#pragma unroll
    for (int a = 0; a < KADIM; ++a)
        s += qgelu_f(p[row * KADIM + a]) * uw[a * KC + c];
    out[idx] += s;
}

// ------------------------------- host driver -------------------------------
extern "C" void kernel_launch(void* const* d_in, const int* in_sizes, int n_in,
                              void* d_out, int out_size, void* d_ws, size_t ws_size,
                              hipStream_t stream)
{
    const float* x        = (const float*)d_in[0];
    const float* pos      = (const float*)d_in[1];
    const float* ln1_g    = (const float*)d_in[2];
    const float* ln1_b    = (const float*)d_in[3];
    const float* qkv_w    = (const float*)d_in[4];
    const float* proj_w   = (const float*)d_in[5];
    const float* proj_b   = (const float*)d_in[6];
    const float* cp1_dw   = (const float*)d_in[7];
    const float* cp1_db   = (const float*)d_in[8];
    const float* cp1_cw   = (const float*)d_in[9];
    const float* cp1_cb   = (const float*)d_in[10];
    const float* cp1_uw   = (const float*)d_in[11];
    const float* cp1_ub   = (const float*)d_in[12];
    const float* ln2_g    = (const float*)d_in[13];
    const float* ln2_b    = (const float*)d_in[14];
    const float* ff_w1    = (const float*)d_in[15];
    const float* ff_b1    = (const float*)d_in[16];
    const float* ff_w2    = (const float*)d_in[17];
    const float* ff_b2    = (const float*)d_in[18];
    const float* cp2_dw   = (const float*)d_in[19];
    const float* cp2_db   = (const float*)d_in[20];
    const float* cp2_cw   = (const float*)d_in[21];
    const float* cp2_cb   = (const float*)d_in[22];
    const float* cp2_uw   = (const float*)d_in[23];
    const float* cp2_ub   = (const float*)d_in[24];
    float* out = (float*)d_out;

    // workspace carve-up
    char* ws = (char*)d_ws;
    constexpr size_t SZ_XRES  = (size_t)KM * KC * 4;        // 32 MB f32
    constexpr size_t SZ_HF    = SZ_XRES;                    // 32 MB f32
    constexpr size_t SZ_HB    = (size_t)KM * KC * 2;        // 16 MB bf16
    constexpr size_t SZ_OB    = SZ_HB;                      // 16 MB bf16
    constexpr size_t SZ_BIG   = (size_t)KM * KMLP * 2;      // 128 MB bf16
    constexpr size_t SZ_D     = (size_t)KM * KADIM * 4;
    constexpr size_t SZ_P     = SZ_D;
    constexpr size_t SZ_WQKV  = (size_t)KC * 3 * KC * 2;
    constexpr size_t SZ_WPROJ = (size_t)KC * KC * 2;
    constexpr size_t SZ_WF1   = (size_t)KC * KMLP * 2;
    constexpr size_t SZ_WF2   = SZ_WF1;

    size_t off = 0;
    float*  x_res   = (float*)(ws + off);  off += SZ_XRES;
    float*  h_f32   = (float*)(ws + off);  off += SZ_HF;
    __bf16* h_bf16  = (__bf16*)(ws + off); off += SZ_HB;
    __bf16* o_bf16  = (__bf16*)(ws + off); off += SZ_OB;
    __bf16* bigbuf  = (__bf16*)(ws + off); off += SZ_BIG;
    float*  dbuf    = (float*)(ws + off);  off += SZ_D;
    float*  pbuf    = (float*)(ws + off);  off += SZ_P;
    __bf16* wt_qkv  = (__bf16*)(ws + off); off += SZ_WQKV;  // [1536,512]
    __bf16* wt_proj = (__bf16*)(ws + off); off += SZ_WPROJ; // [512,512]
    __bf16* wt_ff1  = (__bf16*)(ws + off); off += SZ_WF1;   // [4096,512]
    __bf16* wt_ff2  = (__bf16*)(ws + off); off += SZ_WF2;   // [512,4096]
    (void)off; (void)ws_size; (void)n_in; (void)in_sizes; (void)out_size;

    const int NEL = KM * KC;   // 8388608

    // weight conversions (f32 -> bf16, transposed to [N,K])
    cvt_bf16_t_kernel<<<(KC * 3 * KC + 255) / 256, 256, 0, stream>>>(qkv_w, wt_qkv, KC, 3 * KC);
    cvt_bf16_t_kernel<<<(KC * KC + 255) / 256, 256, 0, stream>>>(proj_w, wt_proj, KC, KC);
    cvt_bf16_t_kernel<<<(KC * KMLP + 255) / 256, 256, 0, stream>>>(ff_w1, wt_ff1, KC, KMLP);
    cvt_bf16_t_kernel<<<(KMLP * KC + 255) / 256, 256, 0, stream>>>(ff_w2, wt_ff2, KMLP, KC);

    // x = x + pos
    add2_kernel<<<NEL / 256, 256, 0, stream>>>(x, pos, x_res, NEL);

    // ---- stage 1: attention + convpass1 ----
    ln_kernel<<<KM, 256, 0, stream>>>(x_res, ln1_g, ln1_b, h_f32, h_bf16);

    // qkv = h @ Wqkv  -> bf16 [KM, 1536]
    gemm_wmma_kernel<0, false, false, true><<<dim3(3 * KC / GT_N, KM / GT_M), 256, 0, stream>>>(
        h_bf16, wt_qkv, nullptr, nullptr, bigbuf, KM, 3 * KC, KC);

    // flash attention -> o_bf16 [KM, 512]
    flash_attn_kernel<<<dim3(KB * KH, KN / 64), 256, 0, stream>>>(bigbuf, o_bf16);

    // x_res += o @ Wproj + proj_b
    gemm_wmma_kernel<0, true, true, false><<<dim3(KC / GT_N, KM / GT_M), 256, 0, stream>>>(
        o_bf16, wt_proj, proj_b, x_res, nullptr, KM, KC, KC);

    // convpass 1 (adds into x_res)
    cp_down_kernel<<<(KM * KADIM) / 256, 256, 0, stream>>>(h_f32, cp1_dw, cp1_db, dbuf);
    cp_conv_kernel<<<(KB * KN * KADIM) / 256, 256, 0, stream>>>(dbuf, cp1_cw, cp1_cb, pbuf);
    cp_up_add_kernel<<<NEL / 256, 256, 0, stream>>>(pbuf, cp1_uw, cp1_ub, x_res);

    // ---- stage 2: FFN + convpass2 ----
    ln_kernel<<<KM, 256, 0, stream>>>(x_res, ln2_g, ln2_b, h_f32, h_bf16);

    // ffn1 = gelu(h @ W1 + b1) -> bf16 [KM, 4096]
    gemm_wmma_kernel<1, false, false, true><<<dim3(KMLP / GT_N, KM / GT_M), 256, 0, stream>>>(
        h_bf16, wt_ff1, ff_b1, nullptr, bigbuf, KM, KMLP, KC);

    // out = x_res;  out += ffn1 @ W2 + b2
    copy_kernel<<<NEL / 256, 256, 0, stream>>>(x_res, out, NEL);
    gemm_wmma_kernel<0, true, true, false><<<dim3(KC / GT_N, KM / GT_M), 256, 0, stream>>>(
        bigbuf, wt_ff2, ff_b2, out, nullptr, KM, KC, KMLP);

    // convpass 2 (adds into out)
    cp_down_kernel<<<(KM * KADIM) / 256, 256, 0, stream>>>(h_f32, cp2_dw, cp2_db, dbuf);
    cp_conv_kernel<<<(KB * KN * KADIM) / 256, 256, 0, stream>>>(dbuf, cp2_cw, cp2_cb, pbuf);
    cp_up_add_kernel<<<NEL / 256, 256, 0, stream>>>(pbuf, cp2_uw, cp2_ub, out);
}
